// Attention_30339648979460
// MI455X (gfx1250) — compile-verified, matrix operand order checked by reading, not run on hardware
//
#include <hip/hip_runtime.h>

typedef _Float16 half_t;
typedef __attribute__((ext_vector_type(16))) _Float16 v16h;
typedef __attribute__((ext_vector_type(8)))  _Float16 v8h;
typedef __attribute__((ext_vector_type(8)))  float    v8f;
typedef __attribute__((ext_vector_type(4)))  float    v4f;

#define ATT_B  4
#define ATT_H  16
#define ATT_S  2048
#define ATT_D  64
#define QTILE  64          // query rows per block (16 per wave, 4 waves)
#define KTILE  32          // keys per inner iteration
#define NWAVES 4
#define NTHREADS (NWAVES * 32)

// softmax in exp2 domain: logits * (1/sqrt(64)) * log2(e)
#define SCALE_LOG2 (0.125f * 1.4426950408889634f)

// load 8 consecutive fp32 (global or LDS), convert to 8 f16
__device__ inline v8h cvt8(const float* p) {
    v4f a = *(const v4f*)(p);
    v4f b = *(const v4f*)(p + 4);
    v8h r;
#pragma unroll
    for (int i = 0; i < 4; ++i) { r[i] = (_Float16)a[i]; r[4 + i] = (_Float16)b[i]; }
    return r;
}

__device__ inline v16h pack16(v8h a, v8h b) {
    v16h r;
#pragma unroll
    for (int i = 0; i < 8; ++i) { r[i] = a[i]; r[8 + i] = b[i]; }
    return r;
}

// CDNA5 async LDS DMA: copy 16 bytes global -> LDS, tracked by ASYNCcnt.
__device__ inline void async_cp128(unsigned lds_off, const float* g) {
    asm volatile("global_load_async_to_lds_b128 %0, %1, off"
                 :: "v"(lds_off), "v"((unsigned long long)(size_t)g)
                 : "memory");
}

__global__ __launch_bounds__(NTHREADS)
void fa_fwd_kernel(const float* __restrict__ Q, const float* __restrict__ K,
                   const float* __restrict__ V, float* __restrict__ O) {
    // Double-buffered raw fp32 landing zones for the async copies,
    // plus f16 compute tiles (K row-major, V transposed) and per-wave P staging.
    __shared__ __align__(16) float  Kraw[2][KTILE * ATT_D];    // 2 x 8 KB
    __shared__ __align__(16) float  Vraw[2][KTILE * ATT_D];    // 2 x 8 KB
    __shared__ __align__(16) half_t Klds[KTILE * ATT_D];       // 4 KB
    __shared__ __align__(16) half_t VldsT[ATT_D * KTILE];      // 4 KB (transposed)
    __shared__ __align__(16) half_t Plds[NWAVES][16 * KTILE];  // 4 KB

    const int tid  = threadIdx.x;
    const int wave = tid >> 5;
    const int lane = tid & 31;
    const int col  = lane & 15;   // column within 16-wide tile
    const int hi   = lane >> 4;   // which 16-lane half

    const int qt = blockIdx.x % (ATT_S / QTILE);
    const int bh = blockIdx.x / (ATT_S / QTILE);
    const int qb = qt * QTILE;
    const int row0 = qb + wave * 16;               // this wave's first query row

    const float* Qb = Q + (size_t)bh * ATT_S * ATT_D;
    const float* Kb = K + (size_t)bh * ATT_S * ATT_D;
    const float* Vb = V + (size_t)bh * ATT_S * ATT_D;
    float*       Ob = O + (size_t)bh * ATT_S * ATT_D;

    // ---- Load Q fragments (A-matrix 16x32 layout, two 32-wide K chunks) ----
    v16h qfrag[2];
#pragma unroll
    for (int c = 0; c < 2; ++c) {
        const float* qrow = Qb + (size_t)(row0 + col) * ATT_D;
        v8h s0 = cvt8(qrow + c * 32 + hi * 8);
        v8h s1 = cvt8(qrow + c * 32 + 16 + hi * 8);
        qfrag[c] = pack16(s0, s1);
    }

    // ---- Flash-attention state ----
    v8f acc[4];
#pragma unroll
    for (int n = 0; n < 4; ++n) acc[n] = (v8f)(0.0f);
    float m_r[8], l_r[8];
#pragma unroll
    for (int r = 0; r < 8; ++r) { m_r[r] = -1.0e30f; l_r[r] = 0.0f; }

    // convert-stage mapping: each thread owns 16 contiguous d of one key row
    const int skey = tid >> 2;
    const int sd   = (tid & 3) * 16;

    const int kend = qb + QTILE;   // causal: keys < qb+64 (>= 2 iterations always)

    // ---- prologue: async-copy tile 0 into buffer 0 ----
#pragma unroll
    for (int i = 0; i < 4; ++i) {
        const int f4 = i * NTHREADS + tid;   // float4 index 0..511 within tile
        async_cp128((unsigned)(size_t)(&Kraw[0][f4 * 4]), Kb + f4 * 4);
        async_cp128((unsigned)(size_t)(&Vraw[0][f4 * 4]), Vb + f4 * 4);
    }

    int cur = 0;
    for (int kb = 0; kb < kend; kb += KTILE) {
        // wait for this wave's outstanding async copies, then block barrier:
        // raw[cur] is complete AND all waves are done reading Klds/VldsT.
        asm volatile("s_wait_asynccnt 0" ::: "memory");
        __syncthreads();

        // ---- kick off async copy of the NEXT tile (overlaps all compute below) ----
        if (kb + KTILE < kend) {
            const int nxt = cur ^ 1;
#pragma unroll
            for (int i = 0; i < 4; ++i) {
                const int f4 = i * NTHREADS + tid;
                async_cp128((unsigned)(size_t)(&Kraw[nxt][f4 * 4]),
                            Kb + (size_t)(kb + KTILE) * ATT_D + f4 * 4);
                async_cp128((unsigned)(size_t)(&Vraw[nxt][f4 * 4]),
                            Vb + (size_t)(kb + KTILE) * ATT_D + f4 * 4);
            }
        }

        // ---- convert landed raw tile: fp32 -> f16, V transposed (LDS->LDS) ----
        {
            const float* kr = &Kraw[cur][skey * ATT_D + sd];
            v8h ka  = cvt8(kr);
            v8h kb2 = cvt8(kr + 8);
            *(v8h*)&Klds[skey * ATT_D + sd]     = ka;
            *(v8h*)&Klds[skey * ATT_D + sd + 8] = kb2;

            const float* vr = &Vraw[cur][skey * ATT_D + sd];
            v8h va  = cvt8(vr);
            v8h vb2 = cvt8(vr + 8);
#pragma unroll
            for (int i = 0; i < 8; ++i) {
                VldsT[(sd + i) * KTILE + skey]     = va[i];
                VldsT[(sd + 8 + i) * KTILE + skey] = vb2[i];
            }
        }
        __syncthreads();

        // ---- S = Q . K^T : two 16x16 key subtiles, contraction 64 = 2 chunks ----
        v8f s[2];
#pragma unroll
        for (int nt = 0; nt < 2; ++nt) {
            v8f cacc = (v8f)(0.0f);
            const int keyl = nt * 16 + col;
#pragma unroll
            for (int c = 0; c < 2; ++c) {
                v8h b0 = *(const v8h*)&Klds[keyl * ATT_D + c * 32 + hi * 8];
                v8h b1 = *(const v8h*)&Klds[keyl * ATT_D + c * 32 + 16 + hi * 8];
                v16h bfrag = pack16(b0, b1);
                cacc = __builtin_amdgcn_wmma_f32_16x16x32_f16(
                    false, qfrag[c], false, bfrag, (short)0, cacc, false, false);
            }
            s[nt] = cacc;
        }

        // ---- online softmax (exp2 domain), per-row stats, stage P to LDS ----
#pragma unroll
        for (int r = 0; r < 8; ++r) {
            const int rowg = row0 + r + hi * 8;
            const int c0 = kb + col;
            const int c1 = c0 + 16;
            float x0 = (c0 <= rowg) ? s[0][r] * SCALE_LOG2 : -1.0e30f;
            float x1 = (c1 <= rowg) ? s[1][r] * SCALE_LOG2 : -1.0e30f;
            float mx = fmaxf(x0, x1);
#pragma unroll
            for (int off = 1; off < 16; off <<= 1)
                mx = fmaxf(mx, __shfl_xor(mx, off, 16));
            float mnew = fmaxf(m_r[r], mx);
            float corr = exp2f(m_r[r] - mnew);
            m_r[r] = mnew;
            float p0 = exp2f(x0 - mnew);
            float p1 = exp2f(x1 - mnew);
            float rs = p0 + p1;
#pragma unroll
            for (int off = 1; off < 16; off <<= 1)
                rs += __shfl_xor(rs, off, 16);
            l_r[r] = l_r[r] * corr + rs;
#pragma unroll
            for (int n = 0; n < 4; ++n) acc[n][r] *= corr;
            const int prow = r + hi * 8;
            Plds[wave][prow * KTILE + col]      = (half_t)p0;
            Plds[wave][prow * KTILE + 16 + col] = (half_t)p1;
        }

        // order per-wave P store -> P load (same-wave LDS ops are in-order,
        // but make the dependency explicit)
        asm volatile("s_wait_dscnt 0" ::: "memory");

        // ---- P A-fragment (16x32) in A-layout ----
        v8h p0 = *(const v8h*)&Plds[wave][col * KTILE + hi * 8];
        v8h p1 = *(const v8h*)&Plds[wave][col * KTILE + 16 + hi * 8];
        v16h pfrag = pack16(p0, p1);

        // ---- O += P . V : 4 Dh chunks of 16; V B-frags contiguous (transposed) ----
#pragma unroll
        for (int n = 0; n < 4; ++n) {
            const int d = n * 16 + col;
            v8h b0 = *(const v8h*)&VldsT[d * KTILE + hi * 8];
            v8h b1 = *(const v8h*)&VldsT[d * KTILE + 16 + hi * 8];
            v16h vfrag = pack16(b0, b1);
            acc[n] = __builtin_amdgcn_wmma_f32_16x16x32_f16(
                false, pfrag, false, vfrag, (short)0, acc[n], false, false);
        }

        cur ^= 1;
    }

    // ---- epilogue: normalize by softmax denominator, store fp32 ----
#pragma unroll
    for (int r = 0; r < 8; ++r) {
        const int rowg = row0 + r + hi * 8;
        const float inv = 1.0f / l_r[r];
#pragma unroll
        for (int n = 0; n < 4; ++n)
            Ob[(size_t)rowg * ATT_D + n * 16 + col] = acc[n][r] * inv;
    }
}

extern "C" void kernel_launch(void* const* d_in, const int* in_sizes, int n_in,
                              void* d_out, int out_size, void* d_ws, size_t ws_size,
                              hipStream_t stream) {
    const float* q = (const float*)d_in[0];
    const float* k = (const float*)d_in[1];
    const float* v = (const float*)d_in[2];
    // d_in[3] is the causal mask; causality is applied analytically in-kernel.
    float* out = (float*)d_out;

    const int blocks = ATT_B * ATT_H * (ATT_S / QTILE);  // 2048
    fa_fwd_kernel<<<dim3(blocks), dim3(NTHREADS), 0, stream>>>(q, k, v, out);
}
